// GVPBranchLigand_15083925143905
// MI455X (gfx1250) — matrix operand order repaired and checked by reading, not compiled, not run on hardware
//
#include <hip/hip_runtime.h>
#include <math.h>

// ---------------------------------------------------------------------------
// GVP-GNN forward for MI455X (gfx1250, wave32).
// All channel-mixing matmuls run through v_wmma_f32_16x16x32_f16.
// Each wave computes a 16x64 output strip: one A fragment shared across 4
// WMMAs per K-step; K loop is software-pipelined two K-steps per iteration
// with ping-pong fragment sets (no register-rotation copies).
// Activations staged as f16 in workspace; f32 accumulation; f32 node state.
// ---------------------------------------------------------------------------

#define EPSF 1e-8f
typedef _Float16 h16;
typedef __attribute__((ext_vector_type(8)))  float    v8f;
typedef __attribute__((ext_vector_type(8)))  _Float16 v8h;
typedef __attribute__((ext_vector_type(16))) _Float16 v16h;

#define GF_RELU 1
#define GF_F16  2
#define GF_ACC  4

// ---------------- generic WMMA GEMM -----------------------------------------
// C[M x Np] = act( A[M x Kp](f16 row-major) * B[Np x Kp](f16, col stored as row) + bias )
// One wave per 16x64 strip. M mult of 16, Kp mult of 32, Np mult of 64.
__global__ __launch_bounds__(32)
void wmma_gemm(const h16* __restrict__ A, const h16* __restrict__ B,
               const float* __restrict__ bias, void* __restrict__ C,
               int M, int Kp, int Np, int Nvalid, int flags)
{
  const int lane = threadIdx.x;
  const int r0 = blockIdx.x * 16;
  const int c0 = blockIdx.y * 64;
  const int row = r0 + (lane & 15);
  const int colb = lane & 15;
  const int kbA = (lane < 16) ? 0 : 16;   // A fragment: 16 contiguous K per lane
  const int kbB = (lane < 16) ? 0 : 16;   // B fragment: 16 contiguous K per lane
  const h16* arow = A + (size_t)row * Kp;
  const h16* bcol = B + (size_t)(c0 + colb) * Kp;
  const size_t bstep = (size_t)16 * Kp;
  v8f acc0 = {}, acc1 = {}, acc2 = {}, acc3 = {};

  // stage-0 fragments: K-step 0  (loop-carried; reloaded in second half)
  v8h  alo0 = *(const v8h*)(arow + kbA);
  v8h  ahi0 = *(const v8h*)(arow + kbA + 8);
  v16h b00  = *(const v16h*)(bcol + kbB);
  v16h b10  = *(const v16h*)(bcol + bstep + kbB);
  v16h b20  = *(const v16h*)(bcol + 2 * bstep + kbB);
  v16h b30  = *(const v16h*)(bcol + 3 * bstep + kbB);

  for (int k0 = 0;;) {
    const int k1 = k0 + 32;
    const bool more1 = k1 < Kp;
    v8h alo1, ahi1; v16h b01, b11, b21, b31;
    if (more1) {                        // stage-1 loads overlap stage-0 WMMAs
      __builtin_prefetch(arow + k1 + 64, 0, 1);   // global_prefetch_b8
      alo1 = *(const v8h*)(arow + k1 + kbA);
      ahi1 = *(const v8h*)(arow + k1 + kbA + 8);
      b01 = *(const v16h*)(bcol + k1 + kbB);
      b11 = *(const v16h*)(bcol + bstep + k1 + kbB);
      b21 = *(const v16h*)(bcol + 2 * bstep + k1 + kbB);
      b31 = *(const v16h*)(bcol + 3 * bstep + k1 + kbB);
    }
    {
      v16h a = __builtin_shufflevector(alo0, ahi0,
                                       0,1,2,3,4,5,6,7,8,9,10,11,12,13,14,15);
      acc0 = __builtin_amdgcn_wmma_f32_16x16x32_f16(false, a, false, b00, (short)0,
                                                    acc0, false, false);
      acc1 = __builtin_amdgcn_wmma_f32_16x16x32_f16(false, a, false, b10, (short)0,
                                                    acc1, false, false);
      acc2 = __builtin_amdgcn_wmma_f32_16x16x32_f16(false, a, false, b20, (short)0,
                                                    acc2, false, false);
      acc3 = __builtin_amdgcn_wmma_f32_16x16x32_f16(false, a, false, b30, (short)0,
                                                    acc3, false, false);
    }
    if (!more1) break;
    const int k2 = k0 + 64;
    const bool more2 = k2 < Kp;
    if (more2) {                        // stage-0 reloads overlap stage-1 WMMAs
      __builtin_prefetch(arow + k2 + 64, 0, 1);
      alo0 = *(const v8h*)(arow + k2 + kbA);
      ahi0 = *(const v8h*)(arow + k2 + kbA + 8);
      b00 = *(const v16h*)(bcol + k2 + kbB);
      b10 = *(const v16h*)(bcol + bstep + k2 + kbB);
      b20 = *(const v16h*)(bcol + 2 * bstep + k2 + kbB);
      b30 = *(const v16h*)(bcol + 3 * bstep + k2 + kbB);
    }
    {
      v16h a = __builtin_shufflevector(alo1, ahi1,
                                       0,1,2,3,4,5,6,7,8,9,10,11,12,13,14,15);
      acc0 = __builtin_amdgcn_wmma_f32_16x16x32_f16(false, a, false, b01, (short)0,
                                                    acc0, false, false);
      acc1 = __builtin_amdgcn_wmma_f32_16x16x32_f16(false, a, false, b11, (short)0,
                                                    acc1, false, false);
      acc2 = __builtin_amdgcn_wmma_f32_16x16x32_f16(false, a, false, b21, (short)0,
                                                    acc2, false, false);
      acc3 = __builtin_amdgcn_wmma_f32_16x16x32_f16(false, a, false, b31, (short)0,
                                                    acc3, false, false);
    }
    if (!more2) break;
    k0 = k2;
  }

  v8f accs[4] = { acc0, acc1, acc2, acc3 };
#pragma unroll
  for (int j = 0; j < 4; ++j) {
    int col = c0 + j * 16 + colb;
    float bval = (bias != nullptr && col < Nvalid) ? bias[col] : 0.0f;
#pragma unroll
    for (int r = 0; r < 8; ++r) {
      int m = r0 + ((lane < 16) ? r : (r + 8));   // C/D layout (ISA 7.12.2)
      size_t idx = (size_t)m * Np + col;
      float v = accs[j][r] + bval;
      if (flags & GF_ACC) v += (flags & GF_F16) ? (float)((const h16*)C)[idx]
                                                : ((const float*)C)[idx];
      if (flags & GF_RELU) v = fmaxf(v, 0.0f);
      if (flags & GF_F16) ((h16*)C)[idx] = (h16)v;
      else                ((float*)C)[idx] = v;
    }
  }
}

// ---------------- small utility kernels -------------------------------------
__global__ void pack_weight(const float* __restrict__ W, int roff, int K, int Nv,
                            int ld, h16* __restrict__ B, int Kp, int Np)
{
  int total = Kp * Np;
  for (int i = blockIdx.x * blockDim.x + threadIdx.x; i < total;
       i += gridDim.x * blockDim.x) {
    int n = i / Kp, k = i % Kp;
    float v = (n < Nv && k < K) ? W[(size_t)(k + roff) * ld + n] : 0.0f;
    B[i] = (h16)v;
  }
}

__global__ void zero_f32(float* p, long n)
{
  for (long i = (long)blockIdx.x * blockDim.x + threadIdx.x; i < n;
       i += (long)gridDim.x * blockDim.x) p[i] = 0.0f;
}

__global__ void f32_to_f16(const float* __restrict__ s, h16* __restrict__ d, long n)
{
  for (long i = (long)blockIdx.x * blockDim.x + threadIdx.x; i < n;
       i += (long)gridDim.x * blockDim.x) d[i] = (h16)s[i];
}

__global__ void count_dst(const int* __restrict__ EI, float* cnt, int E)
{
  int e = blockIdx.x * blockDim.x + threadIdx.x;
  if (e < E) atomicAdd(&cnt[EI[E + e]], 1.0f);
}

__device__ __forceinline__ float block_sum_128(float v, float* red)
{
  int t = threadIdx.x;
  red[t] = v; __syncthreads();
  for (int s = 64; s > 0; s >>= 1) { if (t < s) red[t] += red[t + s]; __syncthreads(); }
  float r = red[0]; __syncthreads();
  return r;
}

// ---------------- input embeddings ------------------------------------------
__global__ void embed_nodes(const float* __restrict__ ns, const float* __restrict__ nv,
                            const float* __restrict__ lnw, const float* __restrict__ lnb,
                            const float* __restrict__ wh, const float* __restrict__ ws,
                            const float* __restrict__ wsb, const float* __restrict__ wv,
                            float* __restrict__ Sn, float* __restrict__ Vn, int N)
{
  __shared__ float red[128];
  __shared__ float sh_s[66];
  __shared__ float sh_nh[64];
  __shared__ float sh_vln[3];
  int n = blockIdx.x, t = threadIdx.x;
  float sv = (t < 66) ? ns[(size_t)n * 66 + t] : 0.0f;
  float mu = block_sum_128(sv, red) / 66.0f;
  float dv = (t < 66) ? (sv - mu) : 0.0f;
  float var = block_sum_128(dv * dv, red) / 66.0f;
  if (t < 66) sh_s[t] = dv * rsqrtf(var + 1e-5f) * lnw[t] + lnb[t];
  if (t == 0) {
    float a = nv[(size_t)n*3], b = nv[(size_t)n*3+1], c = nv[(size_t)n*3+2];
    float rms = sqrtf(fmaxf(a*a + b*b + c*c, EPSF));
    sh_vln[0] = a / rms; sh_vln[1] = b / rms; sh_vln[2] = c / rms;
  }
  __syncthreads();
  if (t < 64) {
    float w = wh[t], s = 0.0f;
    for (int d = 0; d < 3; d++) { float x = sh_vln[d] * w; s += x * x; }
    sh_nh[t] = sqrtf(fmaxf(s, EPSF));
  }
  __syncthreads();
  float acc = wsb[t];
  for (int k = 0; k < 66; k++) acc += sh_s[k] * ws[(size_t)k * 128 + t];
  for (int h = 0; h < 64; h++) acc += sh_nh[h] * ws[(size_t)(66 + h) * 128 + t];
  Sn[(size_t)n * 128 + t] = acc;
  if (t < 64) {
    float coef = 0.0f;
    for (int h = 0; h < 64; h++) coef += wh[h] * wv[(size_t)h * 64 + t];
    for (int d = 0; d < 3; d++) Vn[((size_t)d * N + n) * 64 + t] = sh_vln[d] * coef;
  }
}

__global__ void embed_edges(const float* __restrict__ es, const float* __restrict__ ev,
                            const float* __restrict__ lnw, const float* __restrict__ lnb,
                            const float* __restrict__ wh, const float* __restrict__ ws,
                            const float* __restrict__ wsb, const float* __restrict__ wv,
                            h16* __restrict__ ESh, float* __restrict__ EV, int E)
{
  int e = blockIdx.x * blockDim.x + threadIdx.x;
  if (e >= E) return;
  float s[16]; float mu = 0.0f;
  for (int k = 0; k < 16; k++) { s[k] = es[(size_t)e * 16 + k]; mu += s[k]; }
  mu *= (1.0f / 16.0f);
  float var = 0.0f;
  for (int k = 0; k < 16; k++) { float d = s[k] - mu; var += d * d; }
  var *= (1.0f / 16.0f);
  float rs = rsqrtf(var + 1e-5f);
  for (int k = 0; k < 16; k++) s[k] = (s[k] - mu) * rs * lnw[k] + lnb[k];
  float v0 = ev[(size_t)e*3], v1 = ev[(size_t)e*3+1], v2 = ev[(size_t)e*3+2];
  float rms = sqrtf(fmaxf(v0*v0 + v1*v1 + v2*v2, EPSF));
  float vl[3] = { v0 / rms, v1 / rms, v2 / rms };
  float whv = wh[0];
  float nh = sqrtf(fmaxf((vl[0]*whv)*(vl[0]*whv) + (vl[1]*whv)*(vl[1]*whv)
                         + (vl[2]*whv)*(vl[2]*whv), EPSF));
  for (int o = 0; o < 32; o++) {
    float acc = wsb[o];
    for (int k = 0; k < 16; k++) acc += s[k] * ws[k * 32 + o];
    acc += nh * ws[16 * 32 + o];
    ESh[(size_t)e * 32 + o] = (h16)acc;
  }
  float c = whv * wv[0];
  for (int d = 0; d < 3; d++) EV[(size_t)e * 3 + d] = vl[d] * c;
}

// ---------------- message-layer helpers -------------------------------------
__global__ void compute_mwv(const float* __restrict__ whmid,
                            const float* __restrict__ wv, float* __restrict__ mwv)
{
  int o = threadIdx.x;                 // 64
  float s = 0.0f;
  for (int h = 0; h < 129; h++) s += whmid[h] * wv[h * 64 + o];
  mwv[o] = s;
}

// vn over vh = P[src]+Q[dst]+ev*wh_mid, packed as f16 A (E x 160)
// P/Q planes have row stride 192 (h padded to Np=192).
__global__ void build_asvn(const h16* __restrict__ P, const h16* __restrict__ Q,
                           const float* __restrict__ EV, const int* __restrict__ EI,
                           const float* __restrict__ whmid, h16* __restrict__ AsVn,
                           int E, int N)
{
  int total = E * 160;
  for (int idx = blockIdx.x * blockDim.x + threadIdx.x; idx < total;
       idx += gridDim.x * blockDim.x) {
    int e = idx / 160, k = idx % 160;
    h16 out = (h16)0.0f;
    if (k < 129) {
      int src = EI[e], dst = EI[E + e];
      float wm = whmid[k], s = 0.0f;
      for (int d = 0; d < 3; d++) {
        float x = (float)P[((size_t)d * N + src) * 192 + k]
                + (float)Q[((size_t)d * N + dst) * 192 + k]
                + EV[(size_t)e * 3 + d] * wm;
        s += x * x;
      }
      out = (h16)sqrtf(fmaxf(s, EPSF));
    }
    AsVn[idx] = out;
  }
}

__global__ void msg_s_epilogue(h16* __restrict__ S0h, const h16* __restrict__ U1h,
                               const h16* __restrict__ U3h, const int* __restrict__ EI,
                               const float* __restrict__ wsb, int E)
{
  int total = E * 128;
  for (int idx = blockIdx.x * blockDim.x + threadIdx.x; idx < total;
       idx += gridDim.x * blockDim.x) {
    int e = idx >> 7, j = idx & 127;
    int src = EI[e], dst = EI[E + e];
    float v = (float)S0h[idx] + (float)U1h[(size_t)src * 128 + j]
            + (float)U3h[(size_t)dst * 128 + j] + wsb[j];
    S0h[idx] = (h16)fmaxf(v, 0.0f);
  }
}

__global__ void gate_m0(const h16* __restrict__ Pv, const h16* __restrict__ Qv,
                        const float* __restrict__ EV, const int* __restrict__ EI,
                        const float* __restrict__ mwv, h16* __restrict__ Av,
                        int E, int N)
{
  int total = E * 64;
  for (int idx = blockIdx.x * blockDim.x + threadIdx.x; idx < total;
       idx += gridDim.x * blockDim.x) {
    int e = idx >> 6, o = idx & 63;
    int src = EI[e], dst = EI[E + e];
    float t3[3], s = 0.0f;
    for (int d = 0; d < 3; d++) {
      float x = (float)Pv[((size_t)d * N + src) * 64 + o]
              + (float)Qv[((size_t)d * N + dst) * 64 + o]
              + EV[(size_t)e * 3 + d] * mwv[o];
      t3[d] = x; s += x * x;
    }
    float g = 1.0f / (1.0f + __expf(-sqrtf(fmaxf(s, EPSF))));
    for (int d = 0; d < 3; d++)
      Av[((size_t)d * E + e) * 64 + o] = (h16)(t3[d] * g);
  }
}

// As[m] = [ Sprev(soff) | vn(Hv) | 0 pad ]   (vn from f16 VH planes)
__global__ void pack_post_vh(const h16* __restrict__ VH, int Mrows, int Hp, int Hv,
                             const h16* __restrict__ Sprev, int soff,
                             h16* __restrict__ As, int Kp)
{
  int m = blockIdx.x;
  for (int k = threadIdx.x; k < Kp; k += blockDim.x) {
    h16 val;
    if (k < soff) val = Sprev[(size_t)m * soff + k];
    else if (k < soff + Hv) {
      int h = k - soff; float s = 0.0f;
      for (int d = 0; d < 3; d++) {
        float x = (float)VH[((size_t)d * Mrows + m) * Hp + h];
        s += x * x;
      }
      val = (h16)sqrtf(fmaxf(s, EPSF));
    } else val = (h16)0.0f;
    As[(size_t)m * Kp + k] = val;
  }
}

__global__ void gate_vo(const h16* __restrict__ VO, h16* __restrict__ Av, int M, int C)
{
  int total = M * C;
  for (int idx = blockIdx.x * blockDim.x + threadIdx.x; idx < total;
       idx += gridDim.x * blockDim.x) {
    int m = idx / C, o = idx % C;
    float t3[3], s = 0.0f;
    for (int d = 0; d < 3; d++) {
      float x = (float)VO[((size_t)d * M + m) * C + o];
      t3[d] = x; s += x * x;
    }
    float g = 1.0f / (1.0f + __expf(-sqrtf(fmaxf(s, EPSF))));
    for (int d = 0; d < 3; d++)
      Av[((size_t)d * M + m) * C + o] = (h16)(t3[d] * g);
  }
}

__global__ void aggregate(const h16* __restrict__ MS, const h16* __restrict__ MV,
                          const int* __restrict__ EI, int E, int N,
                          float* __restrict__ AggS, float* __restrict__ AggV)
{
  int e = blockIdx.x, t = threadIdx.x;
  int dst = EI[E + e];
  atomicAdd(&AggS[(size_t)dst * 128 + t], (float)MS[(size_t)e * 128 + t]);
  if (t < 64)
    for (int d = 0; d < 3; d++)
      atomicAdd(&AggV[((size_t)d * N + dst) * 64 + t],
                (float)MV[((size_t)d * E + e) * 64 + t]);
}

// residual + GVP LayerNorm (dS/dV optional, f16 or f32, optional mean by cnt)
__global__ void residual_ln(float* __restrict__ Sn, float* __restrict__ Vn,
                            const void* dS, const void* dV, int df16,
                            const float* __restrict__ cnt,
                            const float* __restrict__ lnw, const float* __restrict__ lnb,
                            int N)
{
  __shared__ float red[128];
  int n = blockIdx.x, t = threadIdx.x;
  float denom = cnt ? fmaxf(cnt[n], 1.0f) : 1.0f;
  size_t si = (size_t)n * 128 + t;
  float sv = Sn[si];
  if (dS) sv += (df16 ? (float)((const h16*)dS)[si] : ((const float*)dS)[si]) / denom;
  float mu = block_sum_128(sv, red) / 128.0f;
  float dvv = sv - mu;
  float var = block_sum_128(dvv * dvv, red) / 128.0f;
  Sn[si] = dvv * rsqrtf(var + 1e-5f) * lnw[t] + lnb[t];
  float vv[3] = {0.f, 0.f, 0.f}; float ss = 0.0f;
  if (t < 64) {
    for (int d = 0; d < 3; d++) {
      size_t vi = ((size_t)d * N + n) * 64 + t;
      float x = Vn[vi];
      if (dV) x += (df16 ? (float)((const h16*)dV)[vi] : ((const float*)dV)[vi]) / denom;
      vv[d] = x; ss += x * x;
    }
    ss = fmaxf(ss, EPSF);
  }
  float tot = block_sum_128((t < 64) ? ss : 0.0f, red);
  float rms = sqrtf(tot / 64.0f);
  if (t < 64)
    for (int d = 0; d < 3; d++)
      Vn[((size_t)d * N + n) * 64 + t] = vv[d] / rms;
}

__global__ void pool_add(const float* __restrict__ OutS, const int* __restrict__ batch,
                         float* __restrict__ pooled, int N)
{
  int n = blockIdx.x, t = threadIdx.x;
  atomicAdd(&pooled[(size_t)batch[n] * 128 + t], OutS[(size_t)n * 128 + t]);
}

// ---------------------------------------------------------------------------
// host orchestration
// ---------------------------------------------------------------------------
// Input pytree (sorted dict keys): 0 node_s, 1 node_v, 2 edge_s, 3 edge_v,
// 4 edge_index, 5 batch, 6 dense.b1, 7 dense.b2, 8 dense.w1, 9 dense.w2,
// 10..13 gvp_e{wh,ws_b,ws_w,wv}, 14..16 gvp_out{wh,ws_b,ws_w},
// 17..20 gvp_v{wh,ws_b,ws_w,wv}, layers l: base=21+24l:
//   +0..3 ff0{wh,ws_b,ws_w,wv}, +4..7 ff1, +8/9 ln0{b,w}, +10/11 ln1{b,w},
//   +12..15 msg0, +16..19 msg1, +20..23 msg2
// 93/94 ln_e{b,w}, 95/96 ln_out{b,w}, 97/98 ln_v{b,w}
extern "C" void kernel_launch(void* const* d_in, const int* in_sizes, int n_in,
                              void* d_out, int out_size, void* d_ws, size_t ws_size,
                              hipStream_t stream)
{
  (void)in_sizes; (void)n_in; (void)out_size; (void)ws_size;
  const int N = 50000, E = 200000, NG = 1024;
  auto F = [&](int i) { return (const float*)d_in[i]; };
  const int* EI    = (const int*)d_in[4];
  const int* BATCH = (const int*)d_in[5];

  size_t cur = 0;
  auto alloc = [&](size_t bytes) -> char* {
    cur = (cur + 255) & ~(size_t)255;
    char* p = (char*)d_ws + cur; cur += bytes; return p;
  };

  // persistent state
  float* Sn     = (float*)alloc((size_t)N * 128 * 4);
  float* Vn     = (float*)alloc((size_t)3 * N * 64 * 4);
  h16*   Snh    = (h16*)  alloc((size_t)N * 128 * 2);
  h16*   ESh    = (h16*)  alloc((size_t)E * 32 * 2);
  float* EV     = (float*)alloc((size_t)E * 3 * 4);
  float* AggS   = (float*)alloc((size_t)N * 128 * 4);
  float* AggV   = (float*)alloc((size_t)3 * N * 64 * 4);
  float* cntp   = (float*)alloc((size_t)N * 4);
  float* pooled = (float*)alloc((size_t)NG * 128 * 4);
  h16*   poolh  = (h16*)  alloc((size_t)NG * 128 * 2);
  h16*   Hh     = (h16*)  alloc((size_t)NG * 256 * 2);

  auto packW = [&](const float* W, int roff, int K, int Nv, int ld,
                   int Kp, int Np) -> h16* {
    h16* B = (h16*)alloc((size_t)Kp * Np * 2);
    int total = Kp * Np;
    pack_weight<<<(total + 255) / 256, 256, 0, stream>>>(W, roff, K, Nv, ld, B, Kp, Np);
    return B;
  };
  auto gemm = [&](const h16* A, const h16* B, const float* bias, void* C,
                  int M, int Kp, int Np, int Nvalid, int flags) {
    dim3 g(M / 16, Np / 64);
    wmma_gemm<<<g, dim3(32), 0, stream>>>(A, B, bias, C, M, Kp, Np, Nvalid, flags);
  };
  auto zero = [&](float* p, long n) {
    zero_f32<<<4096, 256, 0, stream>>>(p, n);
  };
  auto cvt = [&](const float* s, h16* d, long n) {
    f32_to_f16<<<4096, 256, 0, stream>>>(s, d, n);
  };

  // pack all weights (per launch; deterministic)
  h16 *whTop[3], *whBot[3], *W1[3], *W2[3], *W3[3], *W4[3], *wv0p[3];
  h16 *wh1p[3], *ws1p[3], *wv1p[3], *wh2p[3], *ws2p[3], *wv2p[3];
  h16 *fwh0p[3], *fws0p[3], *fwv0p[3], *fwh1p[3], *fws1p[3], *fwv1p[3];
  const float *whmid[3], *wsb0[3], *wsb1[3], *wsb2[3], *fb0[3], *fb1[3];
  const float *ln0w[3], *ln0b[3], *ln1w[3], *ln1b[3];
  float* mwv[3];
  for (int l = 0; l < 3; l++) {
    int base = 21 + 24 * l;
    const float* wh0 = F(base + 12);            // 129x129
    whTop[l] = packW(wh0, 0,  64, 129, 129, 64, 192);
    whBot[l] = packW(wh0, 65, 64, 129, 129, 64, 192);
    whmid[l] = wh0 + 64 * 129;
    const float* ws0 = F(base + 14);            // 417x128
    W1[l] = packW(ws0, 0,   128, 128, 128, 128, 128);
    W2[l] = packW(ws0, 128, 32,  128, 128, 32,  128);
    W3[l] = packW(ws0, 160, 128, 128, 128, 128, 128);
    W4[l] = packW(ws0, 288, 129, 128, 128, 160, 128);
    wsb0[l] = F(base + 13);
    wv0p[l] = packW(F(base + 15), 0, 129, 64, 64, 192, 64);
    mwv[l]  = (float*)alloc(64 * 4);
    compute_mwv<<<1, 64, 0, stream>>>(whmid[l], F(base + 15), mwv[l]);
    wh1p[l] = packW(F(base + 16), 0, 64, 64, 64, 64, 64);
    wsb1[l] = F(base + 17);
    ws1p[l] = packW(F(base + 18), 0, 192, 128, 128, 192, 128);
    wv1p[l] = packW(F(base + 19), 0, 64, 64, 64, 64, 64);
    wh2p[l] = packW(F(base + 20), 0, 64, 64, 64, 64, 64);
    wsb2[l] = F(base + 21);
    ws2p[l] = packW(F(base + 22), 0, 192, 128, 128, 192, 128);
    wv2p[l] = packW(F(base + 23), 0, 64, 64, 64, 64, 64);
    fwh0p[l] = packW(F(base + 0), 0, 64, 128, 128, 64, 128);
    fb0[l]   = F(base + 1);
    fws0p[l] = packW(F(base + 2), 0, 256, 512, 512, 256, 512);
    fwv0p[l] = packW(F(base + 3), 0, 128, 128, 128, 128, 128);
    fwh1p[l] = packW(F(base + 4), 0, 128, 128, 128, 128, 128);
    fb1[l]   = F(base + 5);
    fws1p[l] = packW(F(base + 6), 0, 640, 128, 128, 640, 128);
    fwv1p[l] = packW(F(base + 7), 0, 128, 64, 64, 128, 64);
    ln0b[l] = F(base + 8);  ln0w[l] = F(base + 9);
    ln1b[l] = F(base + 10); ln1w[l] = F(base + 11);
  }
  h16* whOp = packW(F(14), 0, 64, 64, 64, 64, 64);
  const float* wsbO = F(15);
  h16* wsOp = packW(F(16), 0, 192, 128, 128, 192, 128);
  h16* w1p  = packW(F(8), 0, 128, 256, 256, 128, 256);
  h16* w2p  = packW(F(9), 0, 256, 128, 128, 256, 128);
  const float* b1 = F(6); const float* b2 = F(7);

  // scratch arenas (reused across phases)
  char* Xb = alloc(263000000ull);
  char* Fa = alloc(51200000ull);   // msg s0 / s2 / OutS / ds
  char* Fb = alloc(51200000ull);   // msg s1 / ff Sf
  char* AS = alloc(76800000ull);   // concatenated A matrices
  char* G1 = alloc(76800000ull);   // gated v f16 (3,E,64)

  // --- embeddings + degree ---
  zero(cntp, N);
  count_dst<<<(E + 255) / 256, 256, 0, stream>>>(EI, cntp, E);
  embed_nodes<<<N, 128, 0, stream>>>(F(0), F(1), F(98), F(97),
                                     F(17), F(19), F(18), F(20), Sn, Vn, N);
  embed_edges<<<(E + 255) / 256, 256, 0, stream>>>(F(2), F(3), F(94), F(93),
                                     F(10), F(12), F(11), F(13), ESh, EV, E);

  for (int l = 0; l < 3; l++) {
    // ---- msg GVP0 (split node-space + edge-space) ----
    h16* Avn  = (h16*)(Xb + 0);
    h16* P    = (h16*)(Xb + 19200000);   // (3,N,192) f16
    h16* Q    = (h16*)(Xb + 76800000);   // (3,N,192) f16
    h16* Pv   = (h16*)(Xb + 134400000);
    h16* Qv   = (h16*)(Xb + 153600000);
    h16* U1h  = (h16*)(Xb + 172800000);
    h16* U3h  = (h16*)(Xb + 185600000);
    h16* AsVn = (h16*)(Xb + 198400000);
    h16* S0h  = (h16*)Fa;
    h16* S1h  = (h16*)Fb;
    h16* S2h  = (h16*)Fa;
    h16* Asb  = (h16*)AS;
    h16* Av1  = (h16*)G1;

    cvt(Sn, Snh, (long)N * 128);
    cvt(Vn, Avn, (long)3 * N * 64);
    gemm(Avn, whTop[l], nullptr, P,  3 * N, 64, 192, 192, GF_F16);
    gemm(Avn, whBot[l], nullptr, Q,  3 * N, 64, 192, 192, GF_F16);
    gemm(P,   wv0p[l],  nullptr, Pv, 3 * N, 192, 64, 64, GF_F16);
    gemm(Q,   wv0p[l],  nullptr, Qv, 3 * N, 192, 64, 64, GF_F16);
    gemm(Snh, W1[l],    nullptr, U1h, N, 128, 128, 128, GF_F16);
    gemm(Snh, W3[l],    nullptr, U3h, N, 128, 128, 128, GF_F16);
    build_asvn<<<(E * 160 + 255) / 256, 256, 0, stream>>>(P, Q, EV, EI, whmid[l],
                                                          AsVn, E, N);
    gemm(ESh,  W2[l], nullptr, S0h, E, 32,  128, 128, GF_F16);
    gemm(AsVn, W4[l], nullptr, S0h, E, 160, 128, 128, GF_F16 | GF_ACC);
    msg_s_epilogue<<<(E * 128 + 255) / 256, 256, 0, stream>>>(S0h, U1h, U3h, EI,
                                                              wsb0[l], E);
    gate_m0<<<(E * 64 + 255) / 256, 256, 0, stream>>>(Pv, Qv, EV, EI, mwv[l],
                                                      Av1, E, N);
    // ---- msg GVP1 ----
    h16* VH  = (h16*)(Xb + 0);
    h16* VOa = (h16*)(Xb + 76800000);
    gemm(Av1, wh1p[l], nullptr, VH, 3 * E, 64, 64, 64, GF_F16);
    pack_post_vh<<<E, 256, 0, stream>>>(VH, E, 64, 64, S0h, 128, Asb, 192);
    gemm(Asb, ws1p[l], wsb1[l], S1h, E, 192, 128, 128, GF_F16 | GF_RELU);
    gemm(VH,  wv1p[l], nullptr, VOa, 3 * E, 64, 64, 64, GF_F16);
    gate_vo<<<(E * 64 + 255) / 256, 256, 0, stream>>>(VOa, Av1, E, 64);
    // ---- msg GVP2 (no activations) ----
    gemm(Av1, wh2p[l], nullptr, VH, 3 * E, 64, 64, 64, GF_F16);
    pack_post_vh<<<E, 256, 0, stream>>>(VH, E, 64, 64, S1h, 128, Asb, 192);
    gemm(Asb, ws2p[l], wsb2[l], S2h, E, 192, 128, 128, GF_F16);
    gemm(VH,  wv2p[l], nullptr, VOa, 3 * E, 64, 64, 64, GF_F16);
    // ---- mean-aggregate + residual LN ----
    zero(AggS, (long)N * 128);
    zero(AggV, (long)3 * N * 64);
    aggregate<<<E, 128, 0, stream>>>(S2h, VOa, EI, E, N, AggS, AggV);
    residual_ln<<<N, 128, 0, stream>>>(Sn, Vn, AggS, AggV, 0, cntp,
                                       ln0w[l], ln0b[l], N);
    // ---- feed-forward (node space) ----
    h16* fAvn = (h16*)(Xb + 0);
    h16* VHf  = (h16*)(Xb + 19200000);
    h16* VOf  = (h16*)(Xb + 57600000);
    h16* Avf2 = (h16*)(Xb + 96000000);
    h16* VH2f = (h16*)(Xb + 134400000);
    h16* dvf  = (h16*)(Xb + 172800000);
    h16* Sf   = (h16*)Fb;
    h16* dsf  = (h16*)Fa;
    cvt(Vn, fAvn, (long)3 * N * 64);
    cvt(Sn, Snh, (long)N * 128);
    gemm(fAvn, fwh0p[l], nullptr, VHf, 3 * N, 64, 128, 128, GF_F16);
    pack_post_vh<<<N, 256, 0, stream>>>(VHf, N, 128, 128, Snh, 128, Asb, 256);
    gemm(Asb, fws0p[l], fb0[l], Sf, N, 256, 512, 512, GF_F16 | GF_RELU);
    gemm(VHf, fwv0p[l], nullptr, VOf, 3 * N, 128, 128, 128, GF_F16);
    gate_vo<<<(N * 128 + 255) / 256, 256, 0, stream>>>(VOf, Avf2, N, 128);
    gemm(Avf2, fwh1p[l], nullptr, VH2f, 3 * N, 128, 128, 128, GF_F16);
    pack_post_vh<<<N, 256, 0, stream>>>(VH2f, N, 128, 128, Sf, 512, Asb, 640);
    gemm(Asb,  fws1p[l], fb1[l], dsf, N, 640, 128, 128, GF_F16);
    gemm(VH2f, fwv1p[l], nullptr, dvf, 3 * N, 128, 64, 64, GF_F16);
    residual_ln<<<N, 128, 0, stream>>>(Sn, Vn, dsf, dvf, 1, nullptr,
                                       ln1w[l], ln1b[l], N);
  }

  // ---- output head ----
  residual_ln<<<N, 128, 0, stream>>>(Sn, Vn, nullptr, nullptr, 0, nullptr,
                                     F(96), F(95), N);                 // ln_out
  h16* oAvn = (h16*)(Xb + 0);
  h16* VHo  = (h16*)(Xb + 19200000);
  h16* Asb  = (h16*)AS;
  cvt(Vn, oAvn, (long)3 * N * 64);
  cvt(Sn, Snh, (long)N * 128);
  gemm(oAvn, whOp, nullptr, VHo, 3 * N, 64, 64, 64, GF_F16);
  pack_post_vh<<<N, 256, 0, stream>>>(VHo, N, 64, 64, Snh, 128, Asb, 192);
  float* OutS = (float*)Fa;
  gemm(Asb, wsOp, wsbO, OutS, N, 192, 128, 128, GF_RELU);
  zero(pooled, (long)NG * 128);
  pool_add<<<N, 128, 0, stream>>>(OutS, BATCH, pooled, N);
  cvt(pooled, poolh, (long)NG * 128);
  gemm(poolh, w1p, b1, Hh, NG, 128, 256, 256, GF_F16 | GF_RELU);
  gemm(Hh, w2p, b2, (float*)d_out, NG, 256, 128, 128, 0);
}
